// CoupledRNN_19061064860229
// MI455X (gfx1250) — matrix operand order repaired and checked by reading, compiled, not run on hardware
//
#include <hip/hip_runtime.h>

#define B_   256
#define T_   200
#define P1_  1024
#define P2_  512
#define PC_  1536   // P1 + P2
#define O_   512
#define ALPHA1 0.1f
#define ALPHA2 0.05f

typedef __attribute__((ext_vector_type(16))) __bf16 v16bf;
typedef __attribute__((ext_vector_type(8)))  float  v8f;

union BFrag {                 // 32 bytes: 16 bf16 <-> 2x uint4
    uint4 q[2];
    v16bf v;
    unsigned short s[16];
};

__device__ __forceinline__ unsigned short f2bfbits(float f) {
    union { float f; unsigned u; } x; x.f = f;
    unsigned u = x.u;
    return (unsigned short)((u + 0x7FFFu + ((u >> 16) & 1u)) >> 16);   // RNE
}
__device__ __forceinline__ float bfbits2f(unsigned short s) {
    union { unsigned u; float f; } x; x.u = ((unsigned)s) << 16;
    return x.f;
}
// tanh via hardware transcendentals: v_exp_f32 + v_rcp_f32
__device__ __forceinline__ float fast_tanh(float x) {
    float xa = __builtin_fabsf(x);
    float e  = __builtin_amdgcn_exp2f(xa * -2.8853900817779268f);  // e^{-2|x|}
    float t  = (1.0f - e) * __builtin_amdgcn_rcpf(1.0f + e);
    return __builtin_copysignf(t, x);
}

// ---------------------------------------------------------------------------
// Kernel 1: build bf16 W_big (1536x1536), bf16 W_out (512x1536), fused bias,
// and per-unit input-projection weights w0/w1 (I==2).
// ---------------------------------------------------------------------------
__global__ void prep_kernel(const float* __restrict__ W_rec1, const float* __restrict__ W_rec2,
                            const float* __restrict__ W_12,   const float* __restrict__ W_21,
                            const float* __restrict__ W_out,
                            const float* __restrict__ b_in1,  const float* __restrict__ b_in2,
                            const float* __restrict__ b_rec1, const float* __restrict__ b_rec2,
                            const float* __restrict__ b_12,   const float* __restrict__ b_21,
                            const float* __restrict__ W_in1,  const float* __restrict__ W_in2,
                            unsigned short* __restrict__ Wbig, unsigned short* __restrict__ WoutB,
                            float* __restrict__ bias, float* __restrict__ w0, float* __restrict__ w1) {
    const int NW = PC_ * PC_;
    const int NO = O_ * PC_;
    for (int i = blockIdx.x * blockDim.x + threadIdx.x; i < NW + NO + PC_;
         i += gridDim.x * blockDim.x) {
        if (i < NW) {
            int n = i / PC_, k = i % PC_;
            float v;
            if (n < P1_) v = (k < P1_) ? W_rec1[n * P1_ + k] : W_21[n * P2_ + (k - P1_)];
            else {
                int n2 = n - P1_;
                v = (k < P1_) ? W_12[n2 * P1_ + k] : W_rec2[n2 * P2_ + (k - P1_)];
            }
            Wbig[i] = f2bfbits(v);
        } else if (i < NW + NO) {
            WoutB[i - NW] = f2bfbits(W_out[i - NW]);
        } else {
            int n = i - NW - NO;
            if (n < P1_) {
                bias[n] = b_in1[n] + b_rec1[n] + b_21[n];
                w0[n] = W_in1[n * 2 + 0];
                w1[n] = W_in1[n * 2 + 1];
            } else {
                int n2 = n - P1_;
                bias[n] = b_in2[n2] + b_rec2[n2] + b_12[n2];
                w0[n] = W_in2[n2 * 2 + 0];
                w1[n] = W_in2[n2 * 2 + 1];
            }
        }
    }
}

// ---------------------------------------------------------------------------
// Kernel 2: h0[b][n] = place[b][:] . ([W_h1 ; W_h2])[n][:]    (tiny fp32 GEMM)
// ---------------------------------------------------------------------------
__global__ void inith_kernel(const float* __restrict__ place, const float* __restrict__ W_h1,
                             const float* __restrict__ W_h2, float* __restrict__ h0) {
    int idx = blockIdx.x * blockDim.x + threadIdx.x;
    if (idx >= B_ * PC_) return;
    int b = idx / PC_, n = idx % PC_;
    const float4* wr = (const float4*)((n < P1_) ? (W_h1 + (size_t)n * O_)
                                                 : (W_h2 + (size_t)(n - P1_) * O_));
    const float4* pr = (const float4*)(place + (size_t)b * O_);
    float s = 0.f;
    for (int k = 0; k < O_ / 4; ++k) {
        float4 a = pr[k], w = wr[k];
        s += a.x * w.x + a.y * w.y + a.z * w.z + a.w * w.w;
    }
    h0[idx] = s;
}

// ---------------------------------------------------------------------------
// Kernel 3: persistent coupled-RNN recurrence.
// One workgroup per 16-row batch tile (16 WGs). 1024 threads = 32 waves; wave w
// owns output columns [48w, 48w+48) = 3 WMMA N-tiles. h_cat (16x1536, bf16)
// lives in LDS. K-loop is software-pipelined with an explicit ping-pong of two
// fragment sets so loads for chunk k+1 are in flight while WMMAs for chunk k
// execute (avoids s_wait_loadcnt 0 before every WMMA).
// ---------------------------------------------------------------------------
#define LDS_STRIDE (PC_ + 8)   // row byte stride 3088 = 16*193 (keeps b128 aligned)

#define LOAD_A(dst, kk)                                             \
    dst.q[0] = *(const uint4*)&hbuf[lo][(kk) + hi * 8];             \
    dst.q[1] = *(const uint4*)&hbuf[lo][(kk) + 16 + hi * 8];
#define LOAD_B(dst, j, kk) {                                        \
    const unsigned short* _p = brow[j] + (kk) + hi * 8;             \
    dst.q[0] = *(const uint4*)_p; dst.q[1] = *(const uint4*)(_p + 16); }
#define WMMA3(A, B0t, B1t, B2t)                                                                     \
    acc0 = __builtin_amdgcn_wmma_f32_16x16x32_bf16(false, A.v, false, B0t.v, (short)0, acc0, false, false); \
    acc1 = __builtin_amdgcn_wmma_f32_16x16x32_bf16(false, A.v, false, B1t.v, (short)0, acc1, false, false); \
    acc2 = __builtin_amdgcn_wmma_f32_16x16x32_bf16(false, A.v, false, B2t.v, (short)0, acc2, false, false);

__global__ __launch_bounds__(1024) void rnn_kernel(
        const float* __restrict__ inputs, const float* __restrict__ h0,
        const unsigned short* __restrict__ Wbig,
        const float* __restrict__ bias, const float* __restrict__ w0,
        const float* __restrict__ w1,
        float* __restrict__ h1s, float* __restrict__ h2s) {
    __shared__ alignas(16) unsigned short hbuf[16][LDS_STRIDE];
    __shared__ float xin[16][2];

    const int b0   = blockIdx.x * 16;
    const int tid  = threadIdx.x;
    const int wave = tid >> 5;
    const int lane = tid & 31;
    const int lo   = lane & 15;
    const int hi   = lane >> 4;

    // Seed LDS h-state (bf16) from h0.
    for (int i = tid; i < 16 * PC_; i += 1024) {
        int r = i / PC_, c = i % PC_;
        hbuf[r][c] = f2bfbits(h0[(size_t)(b0 + r) * PC_ + c]);
    }
    __syncthreads();

    const int n_base = wave * 48;
    const unsigned short* brow[3];
    float biasv[3], w0v[3], w1v[3];
#pragma unroll
    for (int j = 0; j < 3; ++j) {
        int n = n_base + j * 16 + lo;               // B-matrix row = W_big row n
        brow[j] = Wbig + (size_t)n * PC_;
        biasv[j] = bias[n]; w0v[j] = w0[n]; w1v[j] = w1[n];
    }

    for (int t = 0; t < T_; ++t) {
        if (tid < 32) {                              // stage this step's raw inputs
            int r = tid >> 1, c = tid & 1;
            xin[r][c] = inputs[((size_t)(b0 + r) * T_ + t) * 2 + c];
        }

        v8f acc0 = {}, acc1 = {}, acc2 = {};
        BFrag a0, x00, x01, x02;                     // set 0 (even chunks)
        BFrag a1, x10, x11, x12;                     // set 1 (odd chunks)

        LOAD_A(a0, 0) LOAD_B(x00, 0, 0) LOAD_B(x01, 1, 0) LOAD_B(x02, 2, 0)
        for (int k0 = 0; k0 < PC_ - 64; k0 += 64) {
            LOAD_A(a1, k0 + 32)
            LOAD_B(x10, 0, k0 + 32) LOAD_B(x11, 1, k0 + 32) LOAD_B(x12, 2, k0 + 32)
            WMMA3(a0, x00, x01, x02)
            LOAD_A(a0, k0 + 64)
            LOAD_B(x00, 0, k0 + 64) LOAD_B(x01, 1, k0 + 64) LOAD_B(x02, 2, k0 + 64)
            WMMA3(a1, x10, x11, x12)
        }
        LOAD_A(a1, PC_ - 32)
        LOAD_B(x10, 0, PC_ - 32) LOAD_B(x11, 1, PC_ - 32) LOAD_B(x12, 2, PC_ - 32)
        WMMA3(a0, x00, x01, x02)
        WMMA3(a1, x10, x11, x12)

        __syncthreads();   // all reads of old h done; xin visible

        v8f accs[3] = {acc0, acc1, acc2};
#pragma unroll
        for (int j = 0; j < 3; ++j) {
            int  n    = n_base + j * 16 + lo;
            bool isP1 = (n < P1_);
            float al  = isP1 ? ALPHA1 : ALPHA2;
#pragma unroll
            for (int r = 0; r < 8; ++r) {
                int   m  = r + hi * 8;               // C/D layout: VGPR r, lanes>=16 -> M=r+8
                float tv = accs[j][r] + biasv[j] + xin[m][0] * w0v[j] + xin[m][1] * w1v[j];
                float hp = bfbits2f(hbuf[m][n]);
                float nh = (1.0f - al) * hp + al * fast_tanh(tv);
                hbuf[m][n] = f2bfbits(nh);           // element owned by this lane: no race
                size_t row = (size_t)(b0 + m) * T_ + t;
                if (isP1) h1s[row * P1_ + n]          = nh;
                else      h2s[row * P2_ + (n - P1_)]  = nh;
            }
        }
        __syncthreads();   // new h committed before next step's A reads
    }
}

// ---------------------------------------------------------------------------
// Kernel 4: outputs = [h1s|h2s] @ W_outᵀ.  M = B*T = 51200, N = 512, K = 1536.
// One WG per 16-row M-tile; 8 waves x 4 N-tiles. A loaded fp32 from d_out and
// converted to bf16 in-register; B from L2-resident bf16 W_out.
// ---------------------------------------------------------------------------
__global__ __launch_bounds__(256) void outproj_kernel(
        const float* __restrict__ h1s, const float* __restrict__ h2s,
        const unsigned short* __restrict__ WoutB, float* __restrict__ out) {
    const int m0   = blockIdx.x * 16;
    const int wave = threadIdx.x >> 5;
    const int lane = threadIdx.x & 31;
    const int lo   = lane & 15;
    const int hi   = lane >> 4;
    const int m    = m0 + lo;

    v8f acc[4] = {};
    const unsigned short* brow[4];
#pragma unroll
    for (int j = 0; j < 4; ++j)
        brow[j] = WoutB + (size_t)(wave * 64 + j * 16 + lo) * PC_;

    for (int k0 = 0; k0 < PC_; k0 += 32) {
        const float* abase = (k0 < P1_) ? (h1s + (size_t)m * P1_ + k0)
                                        : (h2s + (size_t)m * P2_ + (k0 - P1_));
        float4 c0 = *(const float4*)(abase + hi * 8);
        float4 c1 = *(const float4*)(abase + hi * 8 + 4);
        float4 c2 = *(const float4*)(abase + 16 + hi * 8);
        float4 c3 = *(const float4*)(abase + 16 + hi * 8 + 4);
        float fv[16] = {c0.x, c0.y, c0.z, c0.w, c1.x, c1.y, c1.z, c1.w,
                        c2.x, c2.y, c2.z, c2.w, c3.x, c3.y, c3.z, c3.w};
        BFrag a;
#pragma unroll
        for (int e = 0; e < 16; ++e) a.s[e] = f2bfbits(fv[e]);

#pragma unroll
        for (int j = 0; j < 4; ++j) {
            BFrag b;
            const unsigned short* p = brow[j] + k0 + hi * 8;
            b.q[0] = *(const uint4*)p; b.q[1] = *(const uint4*)(p + 16);
            acc[j] = __builtin_amdgcn_wmma_f32_16x16x32_bf16(false, a.v, false, b.v, (short)0, acc[j], false, false);
        }
    }
#pragma unroll
    for (int j = 0; j < 4; ++j) {
        int n = wave * 64 + j * 16 + lo;
#pragma unroll
        for (int r = 0; r < 8; ++r)
            out[(size_t)(m0 + r + hi * 8) * O_ + n] = acc[j][r];
    }
}

// ---------------------------------------------------------------------------
extern "C" void kernel_launch(void* const* d_in, const int* in_sizes, int n_in,
                              void* d_out, int out_size, void* d_ws, size_t ws_size,
                              hipStream_t stream) {
    (void)in_sizes; (void)n_in; (void)out_size; (void)ws_size;

    const float* inputs = (const float*)d_in[0];
    const float* place  = (const float*)d_in[1];
    const float* W_in1  = (const float*)d_in[2];
    const float* b_in1  = (const float*)d_in[3];
    const float* W_in2  = (const float*)d_in[4];
    const float* b_in2  = (const float*)d_in[5];
    const float* W_rec1 = (const float*)d_in[6];
    const float* b_rec1 = (const float*)d_in[7];
    const float* W_rec2 = (const float*)d_in[8];
    const float* b_rec2 = (const float*)d_in[9];
    const float* W_12   = (const float*)d_in[10];
    const float* b_12   = (const float*)d_in[11];
    const float* W_21   = (const float*)d_in[12];
    const float* b_21   = (const float*)d_in[13];
    const float* W_out  = (const float*)d_in[14];
    const float* W_h1   = (const float*)d_in[15];
    const float* W_h2   = (const float*)d_in[16];

    char* ws = (char*)d_ws;
    unsigned short* Wbig  = (unsigned short*)ws; ws += (size_t)PC_ * PC_ * 2;
    unsigned short* WoutB = (unsigned short*)ws; ws += (size_t)O_ * PC_ * 2;
    float* bias = (float*)ws; ws += PC_ * 4;
    float* w0   = (float*)ws; ws += PC_ * 4;
    float* w1   = (float*)ws; ws += PC_ * 4;
    float* h0   = (float*)ws; ws += (size_t)B_ * PC_ * 4;

    float* h1s  = (float*)d_out;
    float* h2s  = h1s + (size_t)B_ * T_ * P1_;
    float* outp = h2s + (size_t)B_ * T_ * P2_;

    prep_kernel<<<2048, 256, 0, stream>>>(W_rec1, W_rec2, W_12, W_21, W_out,
                                          b_in1, b_in2, b_rec1, b_rec2, b_12, b_21,
                                          W_in1, W_in2, Wbig, WoutB, bias, w0, w1);
    inith_kernel<<<(B_ * PC_) / 256, 256, 0, stream>>>(place, W_h1, W_h2, h0);
    rnn_kernel<<<B_ / 16, 1024, 0, stream>>>(inputs, h0, Wbig, bias, w0, w1, h1s, h2s);
    outproj_kernel<<<(B_ * T_) / 16, 256, 0, stream>>>(h1s, h2s, WoutB, outp);
}
